// MultiHead_Attention_Block_80616536146223
// MI455X (gfx1250) — matrix operand order repaired and checked by reading, not compiled, not run on hardware
//
#include <hip/hip_runtime.h>
#include <hip/hip_bf16.h>
#include <stdint.h>

#define D_MODEL   1024
#define NUM_HEADS 16
#define D_K       64
#define SEQ       2048
#define BATCH     2
#define MTOT      (BATCH * SEQ)   // 4096 rows in the "flattened" GEMMs

typedef __attribute__((ext_vector_type(16))) __bf16 v16bf;
typedef __attribute__((ext_vector_type(2)))  __bf16 v2bf;
typedef __attribute__((ext_vector_type(8)))  float  v8f;

union Frag16 {
    v16bf v;
    uint4 q[2];
};

// ---- fp32 -> bf16 (RNE); prefer the packed HW convert when the builtin exists
__device__ __forceinline__ uint32_t pack2bf(float a, float b) {
#if __has_builtin(__builtin_amdgcn_cvt_pk_bf16_f32)
    union { v2bf v; uint32_t u; } r;
    r.v = __builtin_amdgcn_cvt_pk_bf16_f32(a, b);
    return r.u;
#else
    union { float f; uint32_t u; } xa, xb;
    xa.f = a; xb.f = b;
    uint32_t ra = (xa.u + 0x7FFFu + ((xa.u >> 16) & 1u)) >> 16;
    uint32_t rb = (xb.u + 0x7FFFu + ((xb.u >> 16) & 1u)) >> 16;
    return (ra & 0xFFFFu) | (rb << 16);
#endif
}

__device__ __forceinline__ unsigned short f2bf(float f) {
    return (unsigned short)pack2bf(f, 0.f);
}

// ---------------------------------------------------------------------------
// GEMM:  Y[m, n] = sum_k X[m, k] * W[n, k] + bias[n]         (Y = X @ W^T + b)
// M = 4096, N = K = 1024.  Block: 256 thr (8 waves), tile 128x128, BK = 32.
// Ping-pong double-buffered LDS: one barrier per K-step, global loads for the
// next tile issued while WMMAs consume the current one.
// MODE 0: write bf16 to [B, H, S, D_K]   (Q / K projections)
// MODE 1: write bf16 to [B, H, D_K, S]   (V projection, transposed)
// MODE 2: write fp32 row-major [M, N]    (output projection -> d_out)
// ---------------------------------------------------------------------------
template <int MODE, bool A_BF16>
__launch_bounds__(256)
__global__ void gemm_wmma(const void* __restrict__ Aptr,
                          const float* __restrict__ W,
                          const float* __restrict__ bias,
                          void* __restrict__ Out) {
    __shared__ __align__(16) unsigned short lsA[2][128 * 40];   // padded: 40 bf16/row
    __shared__ __align__(16) unsigned short lsB[2][128 * 40];

    const int tid   = threadIdx.x;
    const int lane  = tid & 31;
    const int wave  = tid >> 5;
    const int wm    = wave >> 2;            // 0..1  (64 rows each)
    const int wn    = wave & 3;             // 0..3  (32 cols each)
    const int tileM = blockIdx.x * 128;
    const int tileN = blockIdx.y * 128;

    const int half = lane >> 4;             // 0 or 1
    const int mrow = lane & 15;

    const int lrow = tid >> 1;              // 0..127 : LDS row this thread fills
    const int lseg = tid & 1;               // 0..1   : 16-element segment

    v8f acc[4][2] = {};

    // register staging for the in-flight global tile
    uint4  ra0, ra1;       // bf16 A path
    float4 fa[4];          // fp32 A path
    float4 fw[4];          // W (always fp32)

    // ---- issue global loads for K-chunk kk into staging registers ----
    auto issue_loads = [&](int kk) {
        if (A_BF16) {
            const unsigned short* A = (const unsigned short*)Aptr;
            const uint4* src =
                (const uint4*)(A + (size_t)(tileM + lrow) * D_MODEL + kk + lseg * 16);
            ra0 = src[0];
            ra1 = src[1];
        } else {
            const float* A = (const float*)Aptr;
            const float4* src =
                (const float4*)(A + (size_t)(tileM + lrow) * D_MODEL + kk + lseg * 16);
            fa[0] = src[0]; fa[1] = src[1]; fa[2] = src[2]; fa[3] = src[3];
        }
        const float4* wsrc =
            (const float4*)(W + (size_t)(tileN + lrow) * D_MODEL + kk + lseg * 16);
        fw[0] = wsrc[0]; fw[1] = wsrc[1]; fw[2] = wsrc[2]; fw[3] = wsrc[3];
    };

    // ---- convert/stage the in-flight registers into LDS buffer `buf` ----
    auto stage = [&](int buf) {
        unsigned short* dA = &lsA[buf][lrow * 40 + lseg * 16];
        if (A_BF16) {
            *(uint4*)dA       = ra0;
            *(uint4*)(dA + 8) = ra1;
        } else {
            const float* fs = reinterpret_cast<const float*>(fa);
            uint32_t pk[8];
#pragma unroll
            for (int i = 0; i < 8; ++i) pk[i] = pack2bf(fs[2 * i], fs[2 * i + 1]);
            *(uint4*)dA       = *(uint4*)&pk[0];
            *(uint4*)(dA + 8) = *(uint4*)&pk[4];
        }
        {
            const float* fs = reinterpret_cast<const float*>(fw);
            uint32_t pk[8];
#pragma unroll
            for (int i = 0; i < 8; ++i) pk[i] = pack2bf(fs[2 * i], fs[2 * i + 1]);
            unsigned short* dB = &lsB[buf][lrow * 40 + lseg * 16];
            *(uint4*)dB       = *(uint4*)&pk[0];
            *(uint4*)(dB + 8) = *(uint4*)&pk[4];
        }
    };

    issue_loads(0);
    int buf = 0;
    for (int kk = 0; kk < D_MODEL; kk += 32, buf ^= 1) {
        stage(buf);
        if (kk + 32 < D_MODEL) issue_loads(kk + 32);
        __syncthreads();

        const unsigned short* la = lsA[buf];
        const unsigned short* lb = lsB[buf];

        // ---- fragments straight in WMMA VGPR layout ----
        Frag16 af[4], bf2[2];
        const int kbA = half * 16;   // A: lanes 0-15 -> K 0-15 ; lanes 16-31 -> K 16-31
        const int kbB = half * 16;   // B: lanes 0-15 -> K 0-15 ; lanes 16-31 -> K 16-31
#pragma unroll
        for (int mt = 0; mt < 4; ++mt) {
            int r = wm * 64 + mt * 16 + mrow;
            af[mt].q[0] = *(const uint4*)&la[r * 40 + kbA];
            af[mt].q[1] = *(const uint4*)&la[r * 40 + kbA + 8];
        }
#pragma unroll
        for (int nt = 0; nt < 2; ++nt) {
            int r = wn * 32 + nt * 16 + mrow;
            bf2[nt].q[0] = *(const uint4*)&lb[r * 40 + kbB];
            bf2[nt].q[1] = *(const uint4*)&lb[r * 40 + kbB + 8];
        }
#pragma unroll
        for (int mt = 0; mt < 4; ++mt)
#pragma unroll
            for (int nt = 0; nt < 2; ++nt)
                acc[mt][nt] = __builtin_amdgcn_wmma_f32_16x16x32_bf16(
                    false, af[mt].v, false, bf2[nt].v, (short)0, acc[mt][nt], false, false);
        // single barrier per K-step (ping-pong makes the second one unnecessary)
    }

    // ---- epilogue: C layout -> VGPR v holds row v (+8 for lanes 16-31), col = lane&15 ----
#pragma unroll
    for (int nt = 0; nt < 2; ++nt) {
        const int n  = tileN + wn * 32 + nt * 16 + mrow;
        const float bn = bias[n];
#pragma unroll
        for (int mt = 0; mt < 4; ++mt) {
#pragma unroll
            for (int v = 0; v < 8; ++v) {
                const int m   = tileM + wm * 64 + mt * 16 + v + half * 8;
                const float val = acc[mt][nt][v] + bn;
                if (MODE == 2) {
                    ((float*)Out)[(size_t)m * D_MODEL + n] = val;
                } else {
                    const int b = m >> 11, s = m & (SEQ - 1);
                    const int h = n >> 6,  d = n & (D_K - 1);
                    unsigned short* ob = (unsigned short*)Out;
                    if (MODE == 0)
                        ob[((size_t)(b * NUM_HEADS + h) * SEQ + s) * D_K + d] = f2bf(val);
                    else
                        ob[((size_t)(b * NUM_HEADS + h) * D_K + d) * SEQ + s] = f2bf(val);
                }
            }
        }
    }
}

// ---------------------------------------------------------------------------
// Flash attention: one wave (wave32) per 16 query rows of one (b, h).
// 32 keys / iteration: 4 WMMAs for scores, online softmax, 4 WMMAs for P@V.
// Q/K are bf16 [BH, S, 64]; V is bf16 transposed [BH, 64, S].
// Output: bf16 [B*S, 1024] (heads re-interleaved), feeds the final GEMM.
// ---------------------------------------------------------------------------
__launch_bounds__(256)
__global__ void attention_wmma(const unsigned short* __restrict__ Q,
                               const unsigned short* __restrict__ K,
                               const unsigned short* __restrict__ Vt,
                               const int* __restrict__ Mask,
                               unsigned short* __restrict__ Out) {
    __shared__ __align__(16) unsigned short lsP[8 * 16 * 40];  // per-wave P scratch

    const int tid  = threadIdx.x;
    const int lane = tid & 31;
    const int wave = tid >> 5;
    const int gq   = blockIdx.x * 8 + wave;   // global q-tile id, 0..4095
    const int bh   = gq >> 7;                 // (b*H + h)
    const int qt   = gq & 127;
    const int b    = bh >> 4;
    const int h    = bh & 15;
    const int q0   = qt * 16;

    const unsigned short* qp = Q  + (size_t)bh * SEQ * D_K;
    const unsigned short* kp = K  + (size_t)bh * SEQ * D_K;
    const unsigned short* vp = Vt + (size_t)bh * D_K * SEQ;
    const int*            mp = Mask + (size_t)b * SEQ * SEQ;
    unsigned short*       pl = lsP + wave * 16 * 40;

    const int half = lane >> 4;
    const int mrow = lane & 15;

    // Q fragments (A layout), d-chunks c=0 -> d 0..31, c=1 -> d 32..63
    Frag16 qa[2];
    {
        const int kb = half * 16;
#pragma unroll
        for (int c = 0; c < 2; ++c) {
            const unsigned short* base = qp + (size_t)(q0 + mrow) * D_K + c * 32;
            qa[c].q[0] = *(const uint4*)(base + kb);
            qa[c].q[1] = *(const uint4*)(base + kb + 8);
        }
    }

    v8f o[4] = {};                 // 16x64 fp32 output accumulator
    float mrun[8], lrun[8];
#pragma unroll
    for (int v = 0; v < 8; ++v) { mrun[v] = -__builtin_inff(); lrun[v] = 0.f; }

    const float scale = 0.125f;    // 1/sqrt(64)

    for (int j = 0; j < SEQ; j += 32) {
        // prefetch next iteration's K tile and mask rows (emits global_prefetch_b8)
        if (j + 32 < SEQ) {
            __builtin_prefetch(kp + (size_t)(j + 32 + lane) * D_K, 0, 1);
            __builtin_prefetch(mp + (size_t)(q0 + mrow) * SEQ + j + 32, 0, 1);
        }

        // ---- scores: S[16q x 32k] = Q @ K^T  (K rows row-major == B-fragment layout) ----
        v8f s0 = {}, s1 = {};
#pragma unroll
        for (int c = 0; c < 2; ++c) {
            Frag16 kf0, kf1;
            const unsigned short* k0 = kp + (size_t)(j + mrow) * D_K + c * 32 + half * 16;
            const unsigned short* k1 = kp + (size_t)(j + 16 + mrow) * D_K + c * 32 + half * 16;
            kf0.q[0] = *(const uint4*)(k0);
            kf0.q[1] = *(const uint4*)(k0 + 8);
            kf1.q[0] = *(const uint4*)(k1);
            kf1.q[1] = *(const uint4*)(k1 + 8);
            s0 = __builtin_amdgcn_wmma_f32_16x16x32_bf16(false, qa[c].v, false, kf0.v,
                                                         (short)0, s0, false, false);
            s1 = __builtin_amdgcn_wmma_f32_16x16x32_bf16(false, qa[c].v, false, kf1.v,
                                                         (short)0, s1, false, false);
        }

        // ---- mask + scale + online softmax (row reductions across 16-lane halves) ----
        float corr[8];
#pragma unroll
        for (int v = 0; v < 8; ++v) {
            const int row = q0 + v + half * 8;
            const int* mr = mp + (size_t)row * SEQ + j;
            const float a  = mr[mrow]      ? s0[v] * scale : -1e9f;
            const float bb = mr[16 + mrow] ? s1[v] * scale : -1e9f;
            float t = fmaxf(a, bb);
            t = fmaxf(t, __shfl_xor(t, 1));
            t = fmaxf(t, __shfl_xor(t, 2));
            t = fmaxf(t, __shfl_xor(t, 4));
            t = fmaxf(t, __shfl_xor(t, 8));
            const float mn = fmaxf(mrun[v], t);
            const float e0 = __expf(a - mn);
            const float e1 = __expf(bb - mn);
            float rs = e0 + e1;
            rs += __shfl_xor(rs, 1);
            rs += __shfl_xor(rs, 2);
            rs += __shfl_xor(rs, 4);
            rs += __shfl_xor(rs, 8);
            corr[v] = __expf(mrun[v] - mn);
            lrun[v] = lrun[v] * corr[v] + rs;
            mrun[v] = mn;
            // stash P (bf16) into per-wave LDS scratch: one packed convert for both
            const uint32_t pk = pack2bf(e0, e1);
            const int r = v + half * 8;
            pl[r * 40 + mrow]      = (unsigned short)pk;
            pl[r * 40 + 16 + mrow] = (unsigned short)(pk >> 16);
        }
        // rescale running output
#pragma unroll
        for (int d = 0; d < 4; ++d)
#pragma unroll
            for (int v = 0; v < 8; ++v) o[d][v] *= corr[v];

        // cross-lane LDS round trip: wait for DS stores, then read in A layout
        asm volatile("s_wait_dscnt 0" ::: "memory");
        Frag16 pa;
        {
            const int kb = half * 16;
            pa.q[0] = *(const uint4*)&pl[mrow * 40 + kb];
            pa.q[1] = *(const uint4*)&pl[mrow * 40 + kb + 8];
        }

        // ---- O += P @ V : V^T rows are d, contiguous in keys -> B-fragment loads ----
#pragma unroll
        for (int dacc = 0; dacc < 4; ++dacc) {
            Frag16 vf;
            const unsigned short* vb =
                vp + (size_t)(dacc * 16 + mrow) * SEQ + j + half * 16;
            vf.q[0] = *(const uint4*)(vb);
            vf.q[1] = *(const uint4*)(vb + 8);
            o[dacc] = __builtin_amdgcn_wmma_f32_16x16x32_bf16(false, pa.v, false, vf.v,
                                                              (short)0, o[dacc], false, false);
        }
    }

    // ---- finalize: divide by l, write bf16 [B*S, D_MODEL] with heads interleaved ----
    float inv[8];
#pragma unroll
    for (int v = 0; v < 8; ++v) inv[v] = 1.0f / fmaxf(lrun[v], 1e-20f);
#pragma unroll
    for (int dacc = 0; dacc < 4; ++dacc) {
#pragma unroll
        for (int v = 0; v < 8; ++v) {
            const int row = q0 + v + half * 8;
            const int d   = dacc * 16 + mrow;
            Out[(size_t)(b * SEQ + row) * D_MODEL + h * D_K + d] =
                f2bf(o[dacc][v] * inv[v]);
        }
    }
}

// ---------------------------------------------------------------------------
extern "C" void kernel_launch(void* const* d_in, const int* in_sizes, int n_in,
                              void* d_out, int out_size, void* d_ws, size_t ws_size,
                              hipStream_t stream) {
    const float* Query = (const float*)d_in[0];
    const float* Key   = (const float*)d_in[1];
    const float* Vals  = (const float*)d_in[2];
    const int*   Mask  = (const int*)d_in[3];
    const float* W_q   = (const float*)d_in[4];
    const float* b_q   = (const float*)d_in[5];
    const float* W_k   = (const float*)d_in[6];
    const float* b_k   = (const float*)d_in[7];
    const float* W_v   = (const float*)d_in[8];
    const float* b_v   = (const float*)d_in[9];
    const float* W_o   = (const float*)d_in[10];
    const float* b_o   = (const float*)d_in[11];

    // workspace layout: q | k | v^T | attn  (each 4096*1024 bf16 = 8 MB, 32 MB total)
    unsigned short* q_ws = (unsigned short*)d_ws;
    unsigned short* k_ws = q_ws + (size_t)MTOT * D_MODEL;
    unsigned short* v_ws = k_ws + (size_t)MTOT * D_MODEL;
    unsigned short* a_ws = v_ws + (size_t)MTOT * D_MODEL;

    dim3 grid(MTOT / 128, D_MODEL / 128);   // (32, 8)

    gemm_wmma<0, false><<<grid, 256, 0, stream>>>(Query, W_q, b_q, q_ws);
    gemm_wmma<0, false><<<grid, 256, 0, stream>>>(Key,   W_k, b_k, k_ws);
    gemm_wmma<1, false><<<grid, 256, 0, stream>>>(Vals,  W_v, b_v, v_ws);

    // 4096 q-tiles total / 8 waves per block = 512 blocks
    attention_wmma<<<512, 256, 0, stream>>>(q_ws, k_ws, v_ws, Mask, a_ws);

    gemm_wmma<2, true><<<grid, 256, 0, stream>>>(a_ws, W_o, b_o, (float*)d_out);
}
